// TextGuidedDecomposer_3856880631986
// MI455X (gfx1250) — matrix operand order repaired and checked by reading, not compiled, not run on hardware
//
#include <hip/hip_runtime.h>
#include <hip/hip_bf16.h>
#include <math.h>

// ---------------- problem constants ----------------
#define Bc 8
#define Nc 196
#define Dc 512
#define Mc 20
#define Kc 77
#define Hc 8
#define HDc 64
#define KKEEP 51   // max(int(512*0.1),1)

typedef __attribute__((ext_vector_type(16))) __bf16 v16bf;
typedef __attribute__((ext_vector_type(8)))  float  v8f;

struct __align__(16) B16x8 { unsigned int u[4]; };  // 8 bf16 = 16 bytes

// ---------------- WMMA helpers ----------------
__device__ __forceinline__ v8f wmma_bf16(v16bf a, v16bf b, v8f c) {
  return __builtin_amdgcn_wmma_f32_16x16x32_bf16(false, a, false, b, (short)0, c,
                                                 false, false);
}

// A fragment (16x32 bf16, M x K) from an LDS 16x512 row-major tile.
// ISA layout: lanes 0-15 row M=lane hold K {base..base+7, base+16..base+23};
// lanes 16-31 row M=lane-16 hold K {base+8..+15, base+24..+31}.
__device__ __forceinline__ v16bf load_a_frag(const __bf16* ldsA, int lane, int kbase) {
  int row = lane & 15;
  int g   = lane >> 4;
  const __bf16* p0 = ldsA + row * Dc + kbase + g * 8;
  const __bf16* p1 = ldsA + row * Dc + kbase + 16 + g * 8;
  v16bf a;
  ((B16x8*)&a)[0] = *(const B16x8*)p0;
  ((B16x8*)&a)[1] = *(const B16x8*)p1;
  return a;
}

// B fragment (32x16 bf16, K x N) taken from row-major W[N][Kd] (i.e. B = W^T).
// lanes 0-15: col=lane, K={base..base+7, base+16..base+23}; lanes 16-31: col=lane-16, K={base+8..+15, base+24..+31}.
__device__ __forceinline__ v16bf load_b_frag(const __bf16* W, int n0, int Kd,
                                             int lane, int kbase) {
  int col = lane & 15;
  int g   = lane >> 4;
  const __bf16* p = W + (size_t)(n0 + col) * Kd + kbase + g * 8;
  v16bf b;
  ((B16x8*)&b)[0] = *(const B16x8*)p;
  ((B16x8*)&b)[1] = *(const B16x8*)(p + 16);
  return b;
}

// ---------------- small utility kernels ----------------
__global__ __launch_bounds__(256) void zero_f32(float* p, int n) {
  int i = blockIdx.x * 256 + threadIdx.x;
  if (i < n) p[i] = 0.0f;
}

__global__ __launch_bounds__(256) void cvt_bf16(const float* in, __bf16* out, int n) {
  int i = blockIdx.x * 256 + threadIdx.x;
  if (i < n) out[i] = (__bf16)in[i];
}

// mlp_w2 (20x512) -> zero-padded 32x512 bf16
__global__ __launch_bounds__(256) void pack_w2(const float* w2, __bf16* out) {
  int i = blockIdx.x * 256 + threadIdx.x;
  if (i >= 32 * Dc) return;
  int r = i >> 9, c = i & 511;
  out[i] = (r < Mc) ? (__bf16)w2[(size_t)r * Dc + c] : (__bf16)0.0f;
}

// mode 0: LayerNorm -> bf16 ; mode 1: LayerNorm -> f32 ; mode 2: plain f32->bf16
__global__ __launch_bounds__(256) void ln_rows(const float* in, const float* gg,
                                               const float* bb, void* outp,
                                               int mode) {
  int row = blockIdx.x;
  int tid = threadIdx.x;
  __shared__ float red[256];
  const float* x = in + (size_t)row * Dc;
  float a0 = x[tid], a1 = x[tid + 256];
  if (mode == 2) {
    __bf16* o = (__bf16*)outp + (size_t)row * Dc;
    o[tid] = (__bf16)a0; o[tid + 256] = (__bf16)a1;
    return;
  }
  red[tid] = a0 + a1; __syncthreads();
  for (int off = 128; off > 0; off >>= 1) {
    if (tid < off) red[tid] += red[tid + off];
    __syncthreads();
  }
  float mu = red[0] * (1.0f / Dc); __syncthreads();
  float d0 = a0 - mu, d1 = a1 - mu;
  red[tid] = d0 * d0 + d1 * d1; __syncthreads();
  for (int off = 128; off > 0; off >>= 1) {
    if (tid < off) red[tid] += red[tid + off];
    __syncthreads();
  }
  float rstd = rsqrtf(red[0] * (1.0f / Dc) + 1e-5f);
  float y0 = d0 * rstd * gg[tid] + bb[tid];
  float y1 = d1 * rstd * gg[tid + 256] + bb[tid + 256];
  if (mode == 0) {
    __bf16* o = (__bf16*)outp + (size_t)row * Dc;
    o[tid] = (__bf16)y0; o[tid + 256] = (__bf16)y1;
  } else {
    float* o = (float*)outp + (size_t)row * Dc;
    o[tid] = y0; o[tid + 256] = y1;
  }
}

// ---------------- generic WMMA GEMM: C[M x N] = A_bf16[M x 512] @ W_bf16[N x 512]^T + bias ----------------
__global__ __launch_bounds__(256) void gemm_bf16(const __bf16* A, const __bf16* W,
                                                 const float* bias, float* C,
                                                 int Mrows, int Ncols) {
  __shared__ __align__(16) __bf16 Atile[16 * Dc];
  int m0  = blockIdx.x * 16;
  int n00 = blockIdx.y * 128;
  int tid = threadIdx.x;
  for (int idx = tid; idx < 16 * Dc; idx += 256) {
    int r = idx >> 9, c = idx & 511;
    int row = m0 + r;
    Atile[idx] = (row < Mrows) ? A[(size_t)row * Dc + c] : (__bf16)0.0f;
  }
  __syncthreads();
  int w = tid >> 5, lane = tid & 31;
  int n0 = n00 + w * 16;
  v8f acc = {};
  for (int ks = 0; ks < Dc; ks += 32) {
    if (ks + 32 < Dc)
      __builtin_prefetch(W + (size_t)(n0 + (lane & 15)) * Dc + ks + 32, 0, 1);
    v16bf a = load_a_frag(Atile, lane, ks);
    v16bf b = load_b_frag(W, n0, Dc, lane, ks);
    acc = wmma_bf16(a, b, acc);
  }
  int col = n0 + (lane & 15);
  int g = lane >> 4;
  float bv = bias[col];
  for (int r = 0; r < 8; ++r) {
    int row = m0 + r + g * 8;
    if (row < Mrows) C[(size_t)row * Ncols + col] = acc[r] + bv;
  }
}

// ---------------- attention: per (b,h) block ----------------
__global__ __launch_bounds__(256) void attention(const float* qbuf, const float* kbuf,
                                                 const float* vbuf, __bf16* ctx,
                                                 float* attnw) {
  int b = blockIdx.x >> 3;
  int h = blockIdx.x & 7;
  __shared__ float qrow[HDc];
  __shared__ float sc[224];
  __shared__ float red[256];
  int tid = threadIdx.x;
  const float scale = 0.125f;  // 1/sqrt(64)
  for (int kq = 0; kq < Kc; ++kq) {
    if (tid < HDc) qrow[tid] = qbuf[(size_t)kq * Dc + h * HDc + tid];
    __syncthreads();
    float s = -1e30f;
    if (tid < Nc) {
      const float* kp = kbuf + ((size_t)(b * Nc + tid)) * Dc + h * HDc;
      float acc = 0.0f;
      for (int d = 0; d < HDc; ++d) acc += qrow[d] * kp[d];
      s = acc * scale;
    }
    red[tid] = s; __syncthreads();
    for (int off = 128; off > 0; off >>= 1) {
      if (tid < off) red[tid] = fmaxf(red[tid], red[tid + off]);
      __syncthreads();
    }
    float mx = red[0]; __syncthreads();
    float e = 0.0f;
    if (tid < Nc) e = __expf(s - mx);
    red[tid] = e; __syncthreads();
    for (int off = 128; off > 0; off >>= 1) {
      if (tid < off) red[tid] += red[tid + off];
      __syncthreads();
    }
    float inv = 1.0f / red[0]; __syncthreads();
    if (tid < Nc) {
      float a = e * inv;
      sc[tid] = a;
      atomicAdd(&attnw[((size_t)b * Kc + kq) * Nc + tid], a * (1.0f / Hc));
    }
    __syncthreads();
    if (tid < HDc) {
      float acc = 0.0f;
      for (int n2 = 0; n2 < Nc; ++n2)
        acc += sc[n2] * vbuf[((size_t)(b * Nc + n2)) * Dc + h * HDc + tid];
      ctx[((size_t)b * Kc + kq) * Dc + h * HDc + tid] = (__bf16)acc;
    }
    __syncthreads();
  }
}

// ---------------- fused clean -> MLP(W1,LN,gelu) -> W2 -> Q_patch ----------------
// grid: (mt=5, n=196, b=8); each block handles a 16-row panel of the 77xD "clean".
__global__ __launch_bounds__(256) void fuse_mlp(const float* F, const float* attnw,
                                                const float* aoln, const __bf16* W1,
                                                const float* b1, const float* lng,
                                                const float* lnb, const __bf16* W2p,
                                                const float* b2, float* Qpatch) {
  int mt = blockIdx.x, n = blockIdx.y, b = blockIdx.z;
  __shared__ float Frow[Dc];
  __shared__ float svec[Dc];
  __shared__ float wvec[80];
  __shared__ __align__(16) __bf16 Atile[16 * Dc];
  __shared__ float htile[16 * Dc];
  int tid = threadIdx.x;
  const float* Fp = F + ((size_t)(b * Nc + n)) * Dc;
  for (int d = tid; d < Dc; d += 256) Frow[d] = Fp[d];
  for (int k = tid; k < Kc; k += 256) wvec[k] = attnw[((size_t)b * Kc + k) * Nc + n];
  __syncthreads();
  // s[d] = (1/K) * sum_k attn_out_ln[b,k,d] * w[k]
  for (int d = tid; d < Dc; d += 256) {
    float acc = 0.0f;
    const float* ap = aoln + (size_t)b * Kc * Dc + d;
    for (int k = 0; k < Kc; ++k) acc += ap[(size_t)k * Dc] * wvec[k];
    svec[d] = acc * (1.0f / Kc);
  }
  __syncthreads();
  // clean panel (16 rows) as bf16
  for (int idx = tid; idx < 16 * Dc; idx += 256) {
    int r = idx >> 9, d = idx & 511;
    int k = mt * 16 + r;
    float v = 0.0f;
    if (k < Kc)
      v = Frow[d] * (aoln[((size_t)b * Kc + k) * Dc + d] * wvec[k] - svec[d]);
    Atile[idx] = (__bf16)v;
  }
  __syncthreads();
  int w = tid >> 5, lane = tid & 31;
  // GEMM1: h = clean @ W1^T + b1 ; each wave owns 4 column tiles (64 cols)
  {
    v8f c0 = {}, c1 = {}, c2 = {}, c3 = {};
    int n0 = w * 64;
    for (int ks = 0; ks < Dc; ks += 32) {
      v16bf a = load_a_frag(Atile, lane, ks);
      c0 = wmma_bf16(a, load_b_frag(W1, n0,      Dc, lane, ks), c0);
      c1 = wmma_bf16(a, load_b_frag(W1, n0 + 16, Dc, lane, ks), c1);
      c2 = wmma_bf16(a, load_b_frag(W1, n0 + 32, Dc, lane, ks), c2);
      c3 = wmma_bf16(a, load_b_frag(W1, n0 + 48, Dc, lane, ks), c3);
    }
    int g = lane >> 4, colb = lane & 15;
    for (int i = 0; i < 4; ++i) {
      v8f* cp = (i == 0) ? &c0 : (i == 1) ? &c1 : (i == 2) ? &c2 : &c3;
      int col = n0 + i * 16 + colb;
      float bv = b1[col];
      for (int r = 0; r < 8; ++r)
        htile[(r + g * 8) * Dc + col] = (*cp)[r] + bv;
    }
  }
  __syncthreads();
  // row LayerNorm + exact gelu -> bf16 back into Atile (wave w owns rows 2w,2w+1)
  for (int rr = 0; rr < 2; ++rr) {
    int row = w * 2 + rr;
    const float* hp = htile + row * Dc;
    float vals[16];
    float sum = 0.0f;
    for (int j = 0; j < 16; ++j) { vals[j] = hp[lane * 16 + j]; sum += vals[j]; }
    for (int off = 16; off > 0; off >>= 1) sum += __shfl_xor(sum, off, 32);
    float mu = sum * (1.0f / Dc);
    float var = 0.0f;
    for (int j = 0; j < 16; ++j) { float dd = vals[j] - mu; var += dd * dd; }
    for (int off = 16; off > 0; off >>= 1) var += __shfl_xor(var, off, 32);
    float rstd = rsqrtf(var * (1.0f / Dc) + 1e-5f);
    for (int j = 0; j < 16; ++j) {
      int d = lane * 16 + j;
      float x = (vals[j] - mu) * rstd * lng[d] + lnb[d];
      float ge = 0.5f * x * (1.0f + erff(x * 0.70710678118f));
      Atile[row * Dc + d] = (__bf16)ge;
    }
  }
  __syncthreads();
  // GEMM2: comps = h @ W2^T + b2 (N=20 padded to 32: 2 col tiles on waves 0,1)
  if (w < 2) {
    int n0 = w * 16;
    v8f acc = {};
    for (int ks = 0; ks < Dc; ks += 32) {
      v16bf a = load_a_frag(Atile, lane, ks);
      acc = wmma_bf16(a, load_b_frag(W2p, n0, Dc, lane, ks), acc);
    }
    int g = lane >> 4;
    int m = n0 + (lane & 15);
    if (m < Mc) {
      float bv = b2[m];
      for (int r = 0; r < 8; ++r) {
        int kk = mt * 16 + r + g * 8;
        if (kk < Kc) {
          float val = acc[r] + bv;
          atomicAdd(&Qpatch[((size_t)(b * Nc + n)) * Mc + m], val * wvec[kk]);
        }
      }
    }
  }
}

// ---------------- top-k sparsify + normalize (one wave per (b,n,m) row) ----------------
__global__ __launch_bounds__(256) void sparsify(const float* F, const float* tmpl,
                                                const float* Qpatch, float* out) {
  int gid  = blockIdx.x * 8 + (threadIdx.x >> 5);
  int lane = threadIdx.x & 31;
  if (gid >= Bc * Nc * Mc) return;
  int m  = gid % Mc;
  int bn = gid / Mc;
  float c = Qpatch[(size_t)bn * Mc + m];
  const float* Fp = F    + (size_t)bn * Dc + lane * 16;
  const float* Tp = tmpl + (size_t)m  * Dc + lane * 16;
  float q[16];
  unsigned key[16];
  for (int j = 0; j < 16; ++j) {
    q[j]   = Fp[j] * Tp[j] * c;
    key[j] = __float_as_uint(fabsf(q[j]));
  }
  // largest T with count(key >= T) >= KKEEP  == KKEEP-th largest key
  unsigned T = 0u;
  for (int bit = 31; bit >= 0; --bit) {
    unsigned cand = T | (1u << bit);
    int cnt = 0;
    for (int j = 0; j < 16; ++j) cnt += (key[j] >= cand) ? 1 : 0;
    for (int off = 16; off > 0; off >>= 1) cnt += __shfl_xor(cnt, off, 32);
    if (cnt >= KKEEP) T = cand;
  }
  int cnt_gt = 0, cnt_eq = 0;
  for (int j = 0; j < 16; ++j) {
    cnt_gt += (key[j] > T) ? 1 : 0;
    cnt_eq += (key[j] == T) ? 1 : 0;
  }
  int tot_gt = cnt_gt;
  for (int off = 16; off > 0; off >>= 1) tot_gt += __shfl_xor(tot_gt, off, 32);
  int need = KKEEP - tot_gt;  // ties to keep, by increasing d
  int pre = cnt_eq;
  for (int off = 1; off < 32; off <<= 1) {
    int t2 = __shfl_up(pre, off, 32);
    if (lane >= off) pre += t2;
  }
  pre -= cnt_eq;  // exclusive prefix of tie counts
  bool keep[16];
  float ss = 0.0f;
  int tr = pre;
  for (int j = 0; j < 16; ++j) {
    bool kp = key[j] > T;
    if (key[j] == T) { kp = (tr < need); ++tr; }
    keep[j] = kp;
    if (kp) ss += q[j] * q[j];
  }
  for (int off = 16; off > 0; off >>= 1) ss += __shfl_xor(ss, off, 32);
  float inv = 1.0f / fmaxf(sqrtf(ss), 1e-6f);
  float* op = out + ((size_t)bn * Dc + lane * 16) * Mc + m;
  for (int j = 0; j < 16; ++j) op[(size_t)j * Mc] = keep[j] ? q[j] * inv : 0.0f;
}

// ---------------- host-side orchestration ----------------
extern "C" void kernel_launch(void* const* d_in, const int* in_sizes, int n_in,
                              void* d_out, int out_size, void* d_ws, size_t ws_size,
                              hipStream_t stream) {
  (void)in_sizes; (void)n_in; (void)out_size; (void)ws_size;
  const float* F_clean   = (const float*)d_in[0];
  const float* text      = (const float*)d_in[1];
  const float* in_proj_w = (const float*)d_in[2];
  const float* in_proj_b = (const float*)d_in[3];
  const float* out_w     = (const float*)d_in[4];
  const float* out_b     = (const float*)d_in[5];
  const float* ln1_g     = (const float*)d_in[6];
  const float* ln1_b     = (const float*)d_in[7];
  const float* ln2_g     = (const float*)d_in[8];
  const float* ln2_b     = (const float*)d_in[9];
  const float* mlp_w1    = (const float*)d_in[10];
  const float* mlp_b1    = (const float*)d_in[11];
  const float* mlp_ln_g  = (const float*)d_in[12];
  const float* mlp_ln_b  = (const float*)d_in[13];
  const float* mlp_w2    = (const float*)d_in[14];
  const float* mlp_b2    = (const float*)d_in[15];
  const float* templates = (const float*)d_in[16];

  char* ws = (char*)d_ws;
  // workspace layout (byte offsets)
  const size_t off_inproj_bf = 0;                           // 3*512*512 bf16
  const size_t off_outw_bf   = off_inproj_bf + 1572864;     // 512*512 bf16
  const size_t off_w1_bf     = off_outw_bf   + 524288;      // 512*512 bf16
  const size_t off_w2p_bf    = off_w1_bf     + 524288;      // 32*512 bf16
  const size_t off_qln_bf    = off_w2p_bf    + 32768;       // 77*512 bf16
  const size_t off_kln_bf    = off_qln_bf    + 78848;       // 1568*512 bf16
  const size_t off_fbf       = off_kln_bf    + 1605632;     // 1568*512 bf16
  const size_t off_qbuf      = off_fbf       + 1605632;     // 77*512 f32
  const size_t off_kbuf      = off_qbuf      + 157696;      // 1568*512 f32
  const size_t off_vbuf      = off_kbuf      + 3211264;     // 1568*512 f32
  const size_t off_ctx_bf    = off_vbuf      + 3211264;     // 616*512 bf16
  const size_t off_aout      = off_ctx_bf    + 630784;      // 616*512 f32
  const size_t off_aoln      = off_aout      + 1261568;     // 616*512 f32
  const size_t off_qpatch    = off_aoln      + 1261568;     // 8*196*20 f32

  __bf16* inproj_bf = (__bf16*)(ws + off_inproj_bf);
  __bf16* outw_bf   = (__bf16*)(ws + off_outw_bf);
  __bf16* w1_bf     = (__bf16*)(ws + off_w1_bf);
  __bf16* w2p_bf    = (__bf16*)(ws + off_w2p_bf);
  __bf16* qln_bf    = (__bf16*)(ws + off_qln_bf);
  __bf16* kln_bf    = (__bf16*)(ws + off_kln_bf);
  __bf16* f_bf      = (__bf16*)(ws + off_fbf);
  float*  qbuf      = (float*)(ws + off_qbuf);
  float*  kbuf      = (float*)(ws + off_kbuf);
  float*  vbuf      = (float*)(ws + off_vbuf);
  __bf16* ctx_bf    = (__bf16*)(ws + off_ctx_bf);
  float*  aout      = (float*)(ws + off_aout);
  float*  aoln      = (float*)(ws + off_aoln);
  float*  qpatch    = (float*)(ws + off_qpatch);

  float* out_qs    = (float*)d_out;
  float* out_attnw = out_qs + (size_t)Bc * Nc * Dc * Mc;   // 16,056,320 floats in

  // 0) zero accumulators (d_out/ws are poisoned)
  zero_f32<<<(Bc * Kc * Nc + 255) / 256, 256, 0, stream>>>(out_attnw, Bc * Kc * Nc);
  zero_f32<<<(Bc * Nc * Mc + 255) / 256, 256, 0, stream>>>(qpatch, Bc * Nc * Mc);

  // 1) weights -> bf16
  cvt_bf16<<<(3 * Dc * Dc + 255) / 256, 256, 0, stream>>>(in_proj_w, inproj_bf, 3 * Dc * Dc);
  cvt_bf16<<<(Dc * Dc + 255) / 256, 256, 0, stream>>>(out_w, outw_bf, Dc * Dc);
  cvt_bf16<<<(Dc * Dc + 255) / 256, 256, 0, stream>>>(mlp_w1, w1_bf, Dc * Dc);
  pack_w2<<<(32 * Dc + 255) / 256, 256, 0, stream>>>(mlp_w2, w2p_bf);

  // 2) LayerNorm / convert activations
  ln_rows<<<Kc, 256, 0, stream>>>(text, ln1_g, ln1_b, qln_bf, 0);
  ln_rows<<<Bc * Nc, 256, 0, stream>>>(F_clean, ln1_g, ln1_b, kln_bf, 0);
  ln_rows<<<Bc * Nc, 256, 0, stream>>>(F_clean, ln1_g, ln1_b, f_bf, 2);

  // 3) QKV projections (WMMA)
  gemm_bf16<<<dim3((Kc + 15) / 16, Dc / 128), 256, 0, stream>>>(
      qln_bf, inproj_bf, in_proj_b, qbuf, Kc, Dc);
  gemm_bf16<<<dim3(Bc * Nc / 16, Dc / 128), 256, 0, stream>>>(
      kln_bf, inproj_bf + (size_t)Dc * Dc, in_proj_b + Dc, kbuf, Bc * Nc, Dc);
  gemm_bf16<<<dim3(Bc * Nc / 16, Dc / 128), 256, 0, stream>>>(
      f_bf, inproj_bf + (size_t)2 * Dc * Dc, in_proj_b + 2 * Dc, vbuf, Bc * Nc, Dc);

  // 4) attention (+ attn_weights into output tail)
  attention<<<Bc * Hc, 256, 0, stream>>>(qbuf, kbuf, vbuf, ctx_bf, out_attnw);

  // 5) output projection (WMMA) + LN2
  gemm_bf16<<<dim3((Bc * Kc + 15) / 16, Dc / 128), 256, 0, stream>>>(
      ctx_bf, outw_bf, out_b, aout, Bc * Kc, Dc);
  ln_rows<<<Bc * Kc, 256, 0, stream>>>(aout, ln2_g, ln2_b, aoln, 1);

  // 6) fused clean -> MLP -> Q_patch (main WMMA kernel)
  fuse_mlp<<<dim3((Kc + 15) / 16, Nc, Bc), 256, 0, stream>>>(
      F_clean, out_attnw, aoln, w1_bf, mlp_b1, mlp_ln_g, mlp_ln_b, w2p_bf,
      mlp_b2, qpatch);

  // 7) top-k sparsify + normalize into output
  sparsify<<<(Bc * Nc * Mc + 7) / 8, 256, 0, stream>>>(F_clean, templates, qpatch,
                                                       out_qs);
}
